// Bert_lattice_67765993997027
// MI455X (gfx1250) — compile-verified, matrix operand order checked by reading, not run on hardware
//
#include <hip/hip_runtime.h>
#include <hip/hip_bf16.h>

typedef __attribute__((ext_vector_type(16))) _Float16 v16h;
typedef __attribute__((ext_vector_type(8)))  _Float16 v8h;
typedef __attribute__((ext_vector_type(8)))  float    v8f;

#define BB 2
#define SS 256
#define HH 8
#define DD 32
#define TT 256
#define FFD 1024
#define NLAYER 2

// ---------------------------------------------------------------------------
// WMMA helpers (CDNA5 V_WMMA_F32_16X16X32_F16 lane layouts, wave32)
// A operand: 16x32 (MxK) row-major source; lane t reads row (t&15),
//   halves [k0 + (t>>4)*8 .. +8) and [k0+16+(t>>4)*8 .. +8).
// B operand: pass B^T (NxK row-major) with the identical pattern.
// D: lane t holds col n=(t&15); VGPR v holds row m = v + 8*(t>>4).
// ---------------------------------------------------------------------------
__device__ __forceinline__ v16h load_op(const _Float16* rowbase, int k0, int lane) {
  const _Float16* p = rowbase + k0 + ((lane >> 4) << 3);
  v8h lo = *(const v8h*)(p);
  v8h hi = *(const v8h*)(p + 16);
  v16h r;
#pragma unroll
  for (int q = 0; q < 8; ++q) { r[q] = lo[q]; r[8 + q] = hi[q]; }
  return r;
}

__device__ __forceinline__ v8f wmma_step(v16h a, v16h b, v8f c) {
  return __builtin_amdgcn_wmma_f32_16x16x32_f16(false, a, false, b, (short)0, c,
                                                false, false);
}

// ---------------------------------------------------------------------------
// small utility kernels
// ---------------------------------------------------------------------------
__global__ void k_cast(const float* __restrict__ s, _Float16* __restrict__ d, int n) {
  int t = blockIdx.x * 256 + threadIdx.x;
  if (t < n) d[t] = (_Float16)s[t];
}

// src [R][C] f32 -> dst [C][R] f16  (N-major weight copies for B^T operands)
__global__ void k_cast_tr(const float* __restrict__ s, _Float16* __restrict__ d,
                          int R, int C) {
  int t = blockIdx.x * 256 + threadIdx.x;
  if (t < R * C) {
    int r = t / C, c = t % C;
    d[(size_t)c * R + r] = (_Float16)s[t];
  }
}

__global__ void k_zero_h(_Float16* __restrict__ d, int n) {
  int t = blockIdx.x * 256 + threadIdx.x;
  if (t < n) d[t] = (_Float16)0.f;
}

// Bc[b,s,h] = dot(br_h, q[b,s,h,:] + v_param[h,:])
__global__ void k_bc(const float* __restrict__ qf, const float* __restrict__ vparam,
                     const float* __restrict__ brl, float* __restrict__ Bc) {
  int t = blockIdx.x * 256 + threadIdx.x;
  if (t >= BB * SS * HH) return;
  int h = t & 7;
  int bs = t >> 3;
  float s = 0.f;
#pragma unroll
  for (int d = 0; d < DD; ++d)
    s += brl[h * DD + d] * (qf[(size_t)bs * TT + h * DD + d] + vparam[h * DD + d]);
  Bc[t] = s;
}

// build f16 (q+u), (q+v), k, and transposed V [B,H,D,S]
__global__ void k_post_qkv(const float* __restrict__ qf, const float* __restrict__ kf,
                           const float* __restrict__ vf,
                           const float* __restrict__ ul, const float* __restrict__ vl,
                           _Float16* __restrict__ qu16, _Float16* __restrict__ qv16,
                           _Float16* __restrict__ k16, _Float16* __restrict__ valT16) {
  int t = blockIdx.x * 256 + threadIdx.x;
  if (t >= BB * SS * TT) return;
  int hd = t & 255;
  int bs = t >> 8;
  int b = bs >> 8;
  int srow = bs & 255;
  int h = hd >> 5, d = hd & 31;
  float q = qf[t];
  qu16[t] = (_Float16)(q + ul[hd]);
  qv16[t] = (_Float16)(q + vl[hd]);
  k16[t] = (_Float16)kf[t];
  valT16[(((size_t)b * HH + h) * DD + d) * SS + srow] = (_Float16)vf[t];
}

// ---------------------------------------------------------------------------
// generic WMMA GEMM: C[M x N] = A[M x K (lda)] * Bt[N x K (ldb)]^T (+bias,relu)
// block = 256 thr (8 waves), block tile 16(M) x 256(N); grid = (M/16, N/256)
// ---------------------------------------------------------------------------
__global__ __launch_bounds__(256) void k_gemm16(
    const _Float16* __restrict__ A, int lda, const _Float16* __restrict__ Bt, int ldb,
    const float* __restrict__ bias, float* __restrict__ Cf, _Float16* __restrict__ Ch,
    int ldc, int K, int relu) {
  int lane = threadIdx.x & 31;
  int wave = threadIdx.x >> 5;
  int m0 = blockIdx.x * 16;
  int n0 = blockIdx.y * 256 + wave * 32;
  int r = lane & 15;
  const _Float16* arow = A + (size_t)(m0 + r) * lda;
  const _Float16* brow0 = Bt + (size_t)(n0 + r) * ldb;
  const _Float16* brow1 = Bt + (size_t)(n0 + 16 + r) * ldb;
  v8f c0 = {}, c1 = {};
  for (int k = 0; k < K; k += 32) {
    v16h a = load_op(arow, k, lane);
    c0 = wmma_step(a, load_op(brow0, k, lane), c0);
    c1 = wmma_step(a, load_op(brow1, k, lane), c1);
  }
  int n_a = n0 + r, n_b = n_a + 16;
  int mb = m0 + ((lane >> 4) << 3);
  float ba = bias ? bias[n_a] : 0.f;
  float bb = bias ? bias[n_b] : 0.f;
#pragma unroll
  for (int q = 0; q < 8; ++q) {
    int m = mb + q;
    float x0 = c0[q] + ba;
    float x1 = c1[q] + bb;
    if (relu) { x0 = fmaxf(x0, 0.f); x1 = fmaxf(x1, 0.f); }
    if (Cf) { Cf[(size_t)m * ldc + n_a] = x0; Cf[(size_t)m * ldc + n_b] = x1; }
    if (Ch) { Ch[(size_t)m * ldc + n_a] = (_Float16)x0;
              Ch[(size_t)m * ldc + n_b] = (_Float16)x1; }
  }
}

// ---------------------------------------------------------------------------
// rel16[b,i,j,:] = relu( concat(pe[dss],pe[dse],pe[des],pe[dee]) @ Wfus + b )
// block: (b, i, jtile of 16).  A rows gathered from pe16 via index LDS.
// K = 4 segments x 256.  grid.x = B*S*(S/16) = 8192
// ---------------------------------------------------------------------------
__global__ __launch_bounds__(256) void k_rel(const int* __restrict__ pos_s,
                                             const int* __restrict__ pos_e,
                                             const _Float16* __restrict__ pe16,
                                             const _Float16* __restrict__ wfusT,
                                             const float* __restrict__ bfus,
                                             _Float16* __restrict__ rel16) {
  int blk = blockIdx.x;
  int jt = blk & 15;
  int i = (blk >> 4) & 255;
  int b = blk >> 12;
  __shared__ int idx[4][16];
  if (threadIdx.x < 64) {
    int seg = threadIdx.x >> 4, m = threadIdx.x & 15;
    int j = jt * 16 + m;
    int pi = (seg < 2) ? pos_s[b * SS + i] : pos_e[b * SS + i];
    int pj = ((seg & 1) == 0) ? pos_s[b * SS + j] : pos_e[b * SS + j];
    idx[seg][m] = pi - pj + 256;
  }
  __syncthreads();
  int lane = threadIdx.x & 31;
  int wave = threadIdx.x >> 5;
  int n0 = wave * 32;
  int r = lane & 15;
  const _Float16* brow0 = wfusT + (size_t)(n0 + r) * FFD;
  const _Float16* brow1 = wfusT + (size_t)(n0 + 16 + r) * FFD;
  v8f c0 = {}, c1 = {};
  for (int seg = 0; seg < 4; ++seg) {
    const _Float16* arow = pe16 + (size_t)idx[seg][r] * TT;
    for (int kk = 0; kk < 256; kk += 32) {
      v16h a = load_op(arow, kk, lane);
      c0 = wmma_step(a, load_op(brow0, seg * 256 + kk, lane), c0);
      c1 = wmma_step(a, load_op(brow1, seg * 256 + kk, lane), c1);
    }
  }
  int n_a = n0 + r, n_b = n_a + 16;
  size_t rowbase = (((size_t)(b * SS + i)) * SS + jt * 16) * TT;
#pragma unroll
  for (int q = 0; q < 8; ++q) {
    int m = q + ((lane >> 4) << 3);
    rel16[rowbase + (size_t)m * TT + n_a] = (_Float16)fmaxf(c0[q] + bfus[n_a], 0.f);
    rel16[rowbase + (size_t)m * TT + n_b] = (_Float16)fmaxf(c1[q] + bfus[n_b], 0.f);
  }
}

// ---------------------------------------------------------------------------
// AC: score[b,h,i,j] = sum_d (q+u)[b,i,h,d] * k[b,j,h,d]   (K=32, one WMMA)
// grid.x = B*H*(S/16) = 256
// ---------------------------------------------------------------------------
__global__ __launch_bounds__(256) void k_ac(const _Float16* __restrict__ qu16,
                                            const _Float16* __restrict__ k16,
                                            float* __restrict__ score) {
  int blk = blockIdx.x;
  int it = blk & 15;
  int h = (blk >> 4) & 7;
  int b = blk >> 7;
  int lane = threadIdx.x & 31;
  int wave = threadIdx.x >> 5;
  int n0 = wave * 32;
  int r = lane & 15;
  const _Float16* arow = qu16 + (size_t)(b * SS + it * 16 + r) * TT + h * DD;
  const _Float16* brow0 = k16 + (size_t)(b * SS + n0 + r) * TT + h * DD;
  const _Float16* brow1 = k16 + (size_t)(b * SS + n0 + 16 + r) * TT + h * DD;
  v8f c0 = {}, c1 = {};
  v16h a = load_op(arow, 0, lane);
  c0 = wmma_step(a, load_op(brow0, 0, lane), c0);
  c1 = wmma_step(a, load_op(brow1, 0, lane), c1);
  int n_a = n0 + r, n_b = n_a + 16;
  int mbase = it * 16 + ((lane >> 4) << 3);
  float* srow = score + (size_t)(b * HH + h) * SS * SS;
#pragma unroll
  for (int q = 0; q < 8; ++q) {
    int m = mbase + q;
    srow[(size_t)m * SS + n_a] = c0[q];
    srow[(size_t)m * SS + n_b] = c1[q];
  }
}

// ---------------------------------------------------------------------------
// BD (fused r-projection): score[b,h,i,j] += sum_c rel16[b,i,j,c]*P[b,i,h,c] + Bc
// block per (b,i): A = P16[b,i] (16 rows, h padded to 16), B^T rows = rel rows j.
// grid.x = B*S = 512
// ---------------------------------------------------------------------------
__global__ __launch_bounds__(256) void k_bd(const _Float16* __restrict__ P16,
                                            const _Float16* __restrict__ rel16,
                                            const float* __restrict__ Bc,
                                            float* __restrict__ score) {
  int bi = blockIdx.x;
  int b = bi >> 8;
  int i = bi & 255;
  int lane = threadIdx.x & 31;
  int wave = threadIdx.x >> 5;
  int n0 = wave * 32;
  int r = lane & 15;
  const _Float16* arow = P16 + ((size_t)bi * 16 + r) * TT;
  const _Float16* brow0 = rel16 + ((size_t)bi * SS + n0 + r) * TT;
  const _Float16* brow1 = rel16 + ((size_t)bi * SS + n0 + 16 + r) * TT;
  v8f c0 = {}, c1 = {};
  for (int k = 0; k < TT; k += 32) {
    v16h a = load_op(arow, k, lane);
    c0 = wmma_step(a, load_op(brow0, k, lane), c0);
    c1 = wmma_step(a, load_op(brow1, k, lane), c1);
  }
  if (lane < 16) {  // rows m=0..7 are the valid heads
    int n_a = n0 + r, n_b = n_a + 16;
#pragma unroll
    for (int q = 0; q < 8; ++q) {
      float bc = Bc[bi * 8 + q];
      size_t base = ((size_t)(b * HH + q) * SS + i) * SS;
      score[base + n_a] += c0[q] + bc;
      score[base + n_b] += c1[q] + bc;
    }
  }
}

// ---------------------------------------------------------------------------
// softmax over keys (mask j >= len -> -1e15) + attn = prob @ V  (prob in LDS)
// grid.x = B*H*(S/16) = 256
// ---------------------------------------------------------------------------
__global__ __launch_bounds__(256) void k_softmax_attn(
    const float* __restrict__ score, const int* __restrict__ rlen,
    const int* __restrict__ lexn, const _Float16* __restrict__ valT16,
    float* __restrict__ attn) {
  __shared__ __align__(16) _Float16 probLDS[16 * SS];
  __shared__ float red[16][17];
  int blk = blockIdx.x;
  int it = blk & 15;
  int h = (blk >> 4) & 7;
  int b = blk >> 7;
  int limit = rlen[b] + lexn[0];
  const float* srow = score + ((size_t)(b * HH + h) * SS + it * 16) * SS;
  int rr = threadIdx.x >> 4, cc = threadIdx.x & 15;
  float vals[16];
  float mx = -3.0e38f;
#pragma unroll
  for (int jj = 0; jj < 16; ++jj) {
    int j = cc + 16 * jj;
    float sv = (j < limit) ? srow[(size_t)rr * SS + j] : -1.0e15f;
    vals[jj] = sv;
    mx = fmaxf(mx, sv);
  }
  red[rr][cc] = mx;
  __syncthreads();
  float rowmax = red[rr][0];
#pragma unroll
  for (int qq = 1; qq < 16; ++qq) rowmax = fmaxf(rowmax, red[rr][qq]);
  __syncthreads();
  float lsum = 0.f;
#pragma unroll
  for (int jj = 0; jj < 16; ++jj) {
    float e = __expf(vals[jj] - rowmax);
    vals[jj] = e;
    lsum += e;
  }
  red[rr][cc] = lsum;
  __syncthreads();
  float rs = 0.f;
#pragma unroll
  for (int qq = 0; qq < 16; ++qq) rs += red[rr][qq];
  float inv = 1.f / rs;
#pragma unroll
  for (int jj = 0; jj < 16; ++jj)
    probLDS[rr * SS + cc + 16 * jj] = (_Float16)(vals[jj] * inv);
  __syncthreads();
  int wave = threadIdx.x >> 5;
  int lane = threadIdx.x & 31;
  if (wave < 2) {  // wave-uniform: EXEC all ones inside
    int d0 = wave * 16;
    int r = lane & 15;
    const _Float16* arow = probLDS + r * SS;
    const _Float16* brow = valT16 + ((size_t)(b * HH + h) * DD + d0 + r) * SS;
    v8f c = {};
    for (int k = 0; k < SS; k += 32) {
      v16h a = load_op(arow, k, lane);  // ds_load_b128 pair
      c = wmma_step(a, load_op(brow, k, lane), c);
    }
    int d = d0 + r;
    int mbase = it * 16 + ((lane >> 4) << 3);
#pragma unroll
    for (int q = 0; q < 8; ++q) {
      int i = mbase + q;
      attn[((size_t)b * SS + i) * TT + h * DD + d] = c[q];
    }
  }
}

// y = LayerNorm(a + b)  (one row of 256 per block)
__global__ __launch_bounds__(256) void k_addln(const float* __restrict__ a,
                                               const float* __restrict__ bvec,
                                               float* __restrict__ outf,
                                               _Float16* __restrict__ outh) {
  __shared__ float s1[256], s2[256];
  int row = blockIdx.x;
  int t = threadIdx.x;
  float x = a[(size_t)row * TT + t] + bvec[(size_t)row * TT + t];
  s1[t] = x;
  s2[t] = x * x;
  __syncthreads();
  for (int o = 128; o > 0; o >>= 1) {
    if (t < o) { s1[t] += s1[t + o]; s2[t] += s2[t + o]; }
    __syncthreads();
  }
  float mean = s1[0] * (1.f / 256.f);
  float var = s2[0] * (1.f / 256.f) - mean * mean;
  float y = (x - mean) * rsqrtf(var + 1e-5f);
  outf[(size_t)row * TT + t] = y;
  outh[(size_t)row * TT + t] = (_Float16)y;
}

// out = LayerNorm( relu(h16 @ W2 + b2) + y )   fused GEMM+LN, block = 16 rows
__global__ __launch_bounds__(256) void k_ffn2ln(const _Float16* __restrict__ h16,
                                                const _Float16* __restrict__ w2T,
                                                const float* __restrict__ b2,
                                                const float* __restrict__ yf,
                                                float* __restrict__ outf,
                                                _Float16* __restrict__ outh) {
  __shared__ float zl[16 * TT];
  __shared__ float r1[16][17], r2[16][17];
  int m0 = blockIdx.x * 16;
  int lane = threadIdx.x & 31;
  int wave = threadIdx.x >> 5;
  int n0 = wave * 32;
  int r = lane & 15;
  const _Float16* arow = h16 + (size_t)(m0 + r) * FFD;
  const _Float16* brow0 = w2T + (size_t)(n0 + r) * FFD;
  const _Float16* brow1 = w2T + (size_t)(n0 + 16 + r) * FFD;
  v8f c0 = {}, c1 = {};
  for (int k = 0; k < FFD; k += 32) {
    v16h a = load_op(arow, k, lane);
    c0 = wmma_step(a, load_op(brow0, k, lane), c0);
    c1 = wmma_step(a, load_op(brow1, k, lane), c1);
  }
  int n_a = n0 + r, n_b = n_a + 16;
  int mb = (lane >> 4) << 3;
#pragma unroll
  for (int q = 0; q < 8; ++q) {
    int m = mb + q;
    zl[m * TT + n_a] = fmaxf(c0[q] + b2[n_a], 0.f) + yf[(size_t)(m0 + m) * TT + n_a];
    zl[m * TT + n_b] = fmaxf(c1[q] + b2[n_b], 0.f) + yf[(size_t)(m0 + m) * TT + n_b];
  }
  __syncthreads();
  int rr = threadIdx.x >> 4, cc = threadIdx.x & 15;
  float ps = 0.f, ps2 = 0.f;
#pragma unroll
  for (int jj = 0; jj < 16; ++jj) {
    float vz = zl[rr * TT + cc + 16 * jj];
    ps += vz;
    ps2 += vz * vz;
  }
  r1[rr][cc] = ps;
  r2[rr][cc] = ps2;
  __syncthreads();
  float sum = 0.f, sum2 = 0.f;
#pragma unroll
  for (int qq = 0; qq < 16; ++qq) { sum += r1[rr][qq]; sum2 += r2[rr][qq]; }
  float mean = sum * (1.f / 256.f);
  float var = sum2 * (1.f / 256.f) - mean * mean;
  float inv = rsqrtf(var + 1e-5f);
#pragma unroll
  for (int jj = 0; jj < 16; ++jj) {
    int col = cc + 16 * jj;
    float yv = (zl[rr * TT + col] - mean) * inv;
    outf[(size_t)(m0 + rr) * TT + col] = yv;
    outh[(size_t)(m0 + rr) * TT + col] = (_Float16)yv;
  }
}

// ---------------------------------------------------------------------------
extern "C" void kernel_launch(void* const* d_in, const int* in_sizes, int n_in,
                              void* d_out, int out_size, void* d_ws, size_t ws_size,
                              hipStream_t stream) {
  const float* x = (const float*)d_in[0];
  const int* pos_s = (const int*)d_in[1];
  const int* pos_e = (const int*)d_in[2];
  const int* rlen = (const int*)d_in[3];
  const int* lexn = (const int*)d_in[4];
  const float* pe = (const float*)d_in[5];
  const float* W_fus = (const float*)d_in[6];
  const float* b_fus = (const float*)d_in[7];
  const float* Wq = (const float*)d_in[8];
  const float* bq = (const float*)d_in[9];
  const float* Wk = (const float*)d_in[10];
  const float* bk = (const float*)d_in[11];
  const float* Wv = (const float*)d_in[12];
  const float* bv = (const float*)d_in[13];
  const float* Wr = (const float*)d_in[14];
  const float* br = (const float*)d_in[15];
  const float* uP = (const float*)d_in[16];
  const float* vP = (const float*)d_in[17];
  const float* W1 = (const float*)d_in[18];
  const float* b1 = (const float*)d_in[19];
  const float* W2 = (const float*)d_in[20];
  const float* b2 = (const float*)d_in[21];
  (void)in_sizes; (void)n_in; (void)out_size; (void)ws_size;

  char* wsb = (char*)d_ws;
  size_t off = 0;
  auto alloc = [&](size_t bytes) -> void* {
    void* p = wsb + off;
    off = (off + bytes + 255) & ~(size_t)255;
    return p;
  };
  _Float16* pe16 = (_Float16*)alloc((size_t)513 * 256 * 2);
  _Float16* wfusT = (_Float16*)alloc((size_t)256 * 1024 * 2);
  _Float16* wqT = (_Float16*)alloc((size_t)NLAYER * 65536 * 2);
  _Float16* wkT = (_Float16*)alloc((size_t)NLAYER * 65536 * 2);
  _Float16* wvT = (_Float16*)alloc((size_t)NLAYER * 65536 * 2);
  _Float16* wr16 = (_Float16*)alloc((size_t)NLAYER * 65536 * 2);
  _Float16* w1T = (_Float16*)alloc((size_t)NLAYER * 262144 * 2);
  _Float16* w2T = (_Float16*)alloc((size_t)NLAYER * 262144 * 2);
  _Float16* x16 = (_Float16*)alloc((size_t)131072 * 2);
  _Float16* rel16 = (_Float16*)alloc((size_t)BB * SS * SS * TT * 2);
  float* qf = (float*)alloc((size_t)131072 * 4);
  float* kf = (float*)alloc((size_t)131072 * 4);
  float* vf = (float*)alloc((size_t)131072 * 4);
  _Float16* qu16 = (_Float16*)alloc((size_t)131072 * 2);
  _Float16* qv16 = (_Float16*)alloc((size_t)131072 * 2);
  _Float16* k16 = (_Float16*)alloc((size_t)131072 * 2);
  _Float16* valT16 = (_Float16*)alloc((size_t)131072 * 2);
  float* Bc = (float*)alloc((size_t)4096 * 4);
  _Float16* P16 = (_Float16*)alloc((size_t)BB * SS * 16 * TT * 2);
  float* score = (float*)alloc((size_t)BB * HH * SS * SS * 4);
  float* attn = (float*)alloc((size_t)131072 * 4);
  float* yf = (float*)alloc((size_t)131072 * 4);
  _Float16* y16 = (_Float16*)alloc((size_t)131072 * 2);
  float* outf = (float*)alloc((size_t)131072 * 4);
  _Float16* out16 = (_Float16*)alloc((size_t)131072 * 2);
  _Float16* h16 = (_Float16*)alloc((size_t)524288 * 2);

  // ---- weight / input casts (f32 -> f16, transposed to N-major where needed)
  k_cast<<<dim3((513 * 256 + 255) / 256), 256, 0, stream>>>(pe, pe16, 513 * 256);
  k_cast_tr<<<dim3((1024 * 256 + 255) / 256), 256, 0, stream>>>(W_fus, wfusT, 1024, 256);
  for (int l = 0; l < NLAYER; ++l) {
    k_cast_tr<<<dim3(256), 256, 0, stream>>>(Wq + l * 65536, wqT + l * 65536, 256, 256);
    k_cast_tr<<<dim3(256), 256, 0, stream>>>(Wk + l * 65536, wkT + l * 65536, 256, 256);
    k_cast_tr<<<dim3(256), 256, 0, stream>>>(Wv + l * 65536, wvT + l * 65536, 256, 256);
    k_cast<<<dim3(256), 256, 0, stream>>>(Wr + l * 65536, wr16 + l * 65536, 65536);
    k_cast_tr<<<dim3(1024), 256, 0, stream>>>(W1 + l * 262144, w1T + l * 262144, 256, 1024);
    k_cast_tr<<<dim3(1024), 256, 0, stream>>>(W2 + l * 262144, w2T + l * 262144, 1024, 256);
  }
  k_cast<<<dim3(512), 256, 0, stream>>>(x, x16, 131072);
  k_zero_h<<<dim3((BB * SS * 16 * TT + 255) / 256), 256, 0, stream>>>(P16,
                                                                      BB * SS * 16 * TT);

  // ---- rel = relu(four @ Wfus + b)  (stored f16, reused by both layers)
  k_rel<<<dim3(BB * SS * (SS / 16)), 256, 0, stream>>>(pos_s, pos_e, pe16, wfusT,
                                                       b_fus, rel16);

  for (int l = 0; l < NLAYER; ++l) {
    const float* curf = (l == 0) ? x : outf;
    const _Float16* cur16 = (l == 0) ? x16 : out16;
    // q, k, v projections
    k_gemm16<<<dim3(32, 1), 256, 0, stream>>>(cur16, TT, wqT + l * 65536, TT,
                                              bq + l * TT, qf, (_Float16*)nullptr, TT,
                                              TT, 0);
    k_gemm16<<<dim3(32, 1), 256, 0, stream>>>(cur16, TT, wkT + l * 65536, TT,
                                              bk + l * TT, kf, (_Float16*)nullptr, TT,
                                              TT, 0);
    k_gemm16<<<dim3(32, 1), 256, 0, stream>>>(cur16, TT, wvT + l * 65536, TT,
                                              bv + l * TT, vf, (_Float16*)nullptr, TT,
                                              TT, 0);
    k_post_qkv<<<dim3(512), 256, 0, stream>>>(qf, kf, vf, uP + l * TT, vP + l * TT,
                                              qu16, qv16, k16, valT16);
    k_bc<<<dim3(16), 256, 0, stream>>>(qf, vP + l * TT, br + l * TT, Bc);
    // P[b,s,h,c] = (q+v)_h . Wr[c, h*32..]^T   (per-head K=32 GEMM)
    for (int h = 0; h < HH; ++h) {
      k_gemm16<<<dim3(32, 1), 256, 0, stream>>>(
          qv16 + h * DD, TT, wr16 + l * 65536 + h * DD, TT, (const float*)nullptr,
          (float*)nullptr, P16 + h * TT, 16 * TT, DD, 0);
    }
    k_ac<<<dim3(BB * HH * (SS / 16)), 256, 0, stream>>>(qu16, k16, score);
    k_bd<<<dim3(BB * SS), 256, 0, stream>>>(P16, rel16, Bc, score);
    k_softmax_attn<<<dim3(BB * HH * (SS / 16)), 256, 0, stream>>>(score, rlen, lexn,
                                                                  valT16, attn);
    k_addln<<<dim3(BB * SS), 256, 0, stream>>>(curf, attn, yf, y16);
    // FFN
    k_gemm16<<<dim3(32, 4), 256, 0, stream>>>(y16, TT, w1T + l * 262144, TT,
                                              b1 + l * FFD, (float*)nullptr, h16, FFD,
                                              TT, 1);
    float* dstf = (l == NLAYER - 1) ? (float*)d_out : outf;
    k_ffn2ln<<<dim3(32), 256, 0, stream>>>(h16, w2T + l * 262144, b2 + l * TT, yf,
                                           dstf, out16);
  }
}
